// GraphAttentionLayer_69355131895926
// MI455X (gfx1250) — matrix operand order repaired and checked by reading, not compiled, not run on hardware
//
#include <hip/hip_runtime.h>

#define N_NODES 4096
#define DIM     256
#define NH      4
#define DH      64

typedef __bf16 bf16_t;
typedef __attribute__((ext_vector_type(16))) __bf16 v16bf;
typedef __attribute__((ext_vector_type(8)))  __bf16 v8bf;
typedef __attribute__((ext_vector_type(8)))  float  v8f;
typedef __attribute__((ext_vector_type(2)))  float  v2f;

// gfx1250 async global->LDS DMA (ASYNCcnt-tracked). Guarded so the file still
// compiles (manual-copy fallback) on toolchains without these builtins.
#if __has_builtin(__builtin_amdgcn_global_load_async_to_lds_b128) && \
    __has_builtin(__builtin_amdgcn_s_wait_asynccnt)
#define USE_ASYNC_LDS 1
// Builtin signature (from clang diagnostic): arg0 = int4 addrspace(1)* (global
// source), arg1 = int4 addrspace(3)* (LDS dest), arg2/arg3 = imm offset, cpol.
typedef int v4i_vs __attribute__((vector_size(16)));
typedef __attribute__((address_space(1))) v4i_vs* as1_v4i_p;
typedef __attribute__((address_space(3))) v4i_vs* as3_v4i_p;
#else
#define USE_ASYNC_LDS 0
#endif

// A-fragment K-index pattern for 16-bit 16x16x32 WMMA (CDNA5 ISA 7.12.2):
// lane half s = L/16, half-word t in [0,16): K = t%8 + 8*s + 16*(t/8)
__device__ __forceinline__ int a_kpat(int t, int s) {
    return (t & 7) + 8 * s + 16 * (t >> 3);
}

__device__ __forceinline__ v16bf cat8(v8bf lo, v8bf hi) {
    return __builtin_shufflevector(lo, hi, 0,1,2,3,4,5,6,7,8,9,10,11,12,13,14,15);
}

// ---------------------------------------------------------------------------
// Kernel 1: fused QKV projection.  One block = one 16-row tile of x.
// f32 WMMA 16x16x4 keeps q_proj (residual path) near fp32-exact.
// Writes q_proj fp32 and Q/K/V in bf16 for the attention kernel.
// ---------------------------------------------------------------------------
__global__ __launch_bounds__(256) void qkv_kernel(
    const float* __restrict__ x,
    const float* __restrict__ Wq, const float* __restrict__ bq,
    const float* __restrict__ Wk, const float* __restrict__ bk,
    const float* __restrict__ Wv, const float* __restrict__ bv,
    float* __restrict__ qproj, bf16_t* __restrict__ Qb,
    bf16_t* __restrict__ Kb, bf16_t* __restrict__ Vb)
{
    __shared__ float xs[16][260];                  // pad 260 -> conflict-free col reads
    const int t = threadIdx.x;
    const int rowBase = blockIdx.x * 16;
    for (int i = t; i < 16 * 256; i += 256) {
        int r = i >> 8, c = i & 255;
        xs[r][c] = x[(rowBase + r) * DIM + c];
    }
    __syncthreads();

    const int wave = t >> 5, lane = t & 31, lm = lane & 15, ls = lane >> 4;
    // 48 (matrix, col-tile) pairs over 8 waves -> 6 each
    for (int q = 0; q < 6; ++q) {
        int p = wave * 6 + q;
        int mat = p >> 4, ct = p & 15, colBase = ct * 16;
        const float* W    = (mat == 0) ? Wq : (mat == 1) ? Wk : Wv;
        const float* bias = (mat == 0) ? bq : (mat == 1) ? bk : bv;
        v8f acc = {0.f, 0.f, 0.f, 0.f, 0.f, 0.f, 0.f, 0.f};
        #pragma unroll 8
        for (int k0 = 0; k0 < DIM; k0 += 4) {
            int k = k0 + 2 * ls;                  // f32 frag: vreg j -> K = k + j
            v2f a, b;
            a.x = xs[lm][k];
            a.y = xs[lm][k + 1];
            b.x = W[k * DIM + colBase + lm];
            b.y = W[(k + 1) * DIM + colBase + lm];
            acc = __builtin_amdgcn_wmma_f32_16x16x4_f32(
                false, a, false, b, (short)0, acc, false, false);
        }
        int col = colBase + lm;
        float bb = bias[col];
        #pragma unroll
        for (int r = 0; r < 8; ++r) {
            int row = rowBase + r + 8 * ls;       // C/D layout
            float val = acc[r] + bb;
            int idx = row * DIM + col;
            if (mat == 0)      { qproj[idx] = val; Qb[idx] = (bf16_t)val; }
            else if (mat == 1) { Kb[idx] = (bf16_t)val; }
            else               { Vb[idx] = (bf16_t)val; }
        }
    }
}

// ---------------------------------------------------------------------------
// Kernel 2: masked flash attention.  Block = 8 waves = 2 row-subtiles x 4 heads
// (32 rows per block); m streamed in chunks of 32 with online softmax.
// bf16 WMMA 16x16x32 with f32 accumulation for both S = Q.K^T and P.V.
// K/V chunks are double-buffered and prefetched with async global->LDS DMA.
// ---------------------------------------------------------------------------
__global__ __launch_bounds__(256) void attn_kernel(
    const bf16_t* __restrict__ Qb, const bf16_t* __restrict__ Kb,
    const bf16_t* __restrict__ Vb, const int* __restrict__ adj,
    float* __restrict__ hbuf)
{
    __shared__ bf16_t Ks[2][32][264];             // double-buffered K chunk (16.5 KB ea)
    __shared__ bf16_t Vs[2][32][264];             // double-buffered V chunk
    __shared__ bf16_t Ps[8][16][40];              // per-wave P tile; 80B pitch (16B-aligned)

    const int t = threadIdx.x;
    const int wave = t >> 5, lane = t & 31, lm = lane & 15, ls = lane >> 4;
    const int rt = wave & 1, h = wave >> 1;
    const int rowBase = blockIdx.x * 32 + rt * 16;
    const float NEG_INF = -__builtin_inff();

    auto stage = [&](int bufi, int mbase) {
#if USE_ASYNC_LDS
        // 32x256 bf16 per matrix = 1024 x 16B DMA transfers; 4 per thread each.
        for (int i = t; i < 1024; i += 256) {
            int r = i >> 5, c8 = (i & 31) << 3;
            __builtin_amdgcn_global_load_async_to_lds_b128(
                (as1_v4i_p)(Kb + (size_t)(mbase + r) * DIM + c8),
                (as3_v4i_p)&Ks[bufi][r][c8], 0, 0);
            __builtin_amdgcn_global_load_async_to_lds_b128(
                (as1_v4i_p)(Vb + (size_t)(mbase + r) * DIM + c8),
                (as3_v4i_p)&Vs[bufi][r][c8], 0, 0);
        }
#else
        for (int i = t; i < 32 * 256; i += 256) {
            int r = i >> 8, c = i & 255;
            Ks[bufi][r][c] = Kb[(mbase + r) * DIM + c];
            Vs[bufi][r][c] = Vb[(mbase + r) * DIM + c];
        }
#endif
    };

    // Q A-fragments (two 32-wide d chunks), resident in VGPRs for entire loop
    v16bf aq0, aq1;
    #pragma unroll
    for (int tt = 0; tt < 16; ++tt) {
        int k = 16 * ls + tt;
        aq0[tt] = Qb[(rowBase + lm) * DIM + h * DH + k];
        aq1[tt] = Qb[(rowBase + lm) * DIM + h * DH + 32 + k];
    }

    float m_i[8], l_i[8];
    v8f accv[4];                                  // 16 rows x 64 d, as 4 C-fragments
    #pragma unroll
    for (int r = 0; r < 8; ++r) { m_i[r] = NEG_INF; l_i[r] = 0.f; }
    #pragma unroll
    for (int j = 0; j < 4; ++j) accv[j] = (v8f){0.f,0.f,0.f,0.f,0.f,0.f,0.f,0.f};

    stage(0, 0);
#if USE_ASYNC_LDS
    __builtin_amdgcn_s_wait_asynccnt(0);
#endif
    __syncthreads();

    for (int mb = 0; mb < N_NODES; mb += 32) {
        const int cur = (mb >> 5) & 1;
        if (mb + 32 < N_NODES) stage(cur ^ 1, mb + 32);   // prefetch next chunk

        // S tiles for m columns [mb, mb+16) and [mb+16, mb+32)
        v8f s0 = (v8f){0.f,0.f,0.f,0.f,0.f,0.f,0.f,0.f};
        v8f s1 = s0;
        {
            const int base = h * DH + 16 * ls;    // multiple of 16 halfs (32B aligned)
            v16bf b;
            b = cat8(*(const v8bf*)&Ks[cur][lm][base],
                     *(const v8bf*)&Ks[cur][lm][base + 8]);
            s0 = __builtin_amdgcn_wmma_f32_16x16x32_bf16(false, aq0, false, b, (short)0, s0, false, false);
            b = cat8(*(const v8bf*)&Ks[cur][lm][base + 32],
                     *(const v8bf*)&Ks[cur][lm][base + 40]);
            s0 = __builtin_amdgcn_wmma_f32_16x16x32_bf16(false, aq1, false, b, (short)0, s0, false, false);
            b = cat8(*(const v8bf*)&Ks[cur][16 + lm][base],
                     *(const v8bf*)&Ks[cur][16 + lm][base + 8]);
            s1 = __builtin_amdgcn_wmma_f32_16x16x32_bf16(false, aq0, false, b, (short)0, s1, false, false);
            b = cat8(*(const v8bf*)&Ks[cur][16 + lm][base + 32],
                     *(const v8bf*)&Ks[cur][16 + lm][base + 40]);
            s1 = __builtin_amdgcn_wmma_f32_16x16x32_bf16(false, aq1, false, b, (short)0, s1, false, false);
        }

        // mask + online softmax, per C-fragment row r (row n = r + 8*ls)
        #pragma unroll
        for (int r = 0; r < 8; ++r) {
            int gn = rowBase + r + 8 * ls;
            float v0 = s0[r] * 0.125f;            // 1/sqrt(64)
            float v1 = s1[r] * 0.125f;
            if (adj[gn * N_NODES + mb + lm] == 0)      v0 = NEG_INF;
            if (adj[gn * N_NODES + mb + 16 + lm] == 0) v1 = NEG_INF;
            float cm = fmaxf(v0, v1);
            #pragma unroll
            for (int m = 1; m < 16; m <<= 1) cm = fmaxf(cm, __shfl_xor(cm, m));
            float mn = fmaxf(m_i[r], cm);
            float alpha, p0, p1;
            if (mn == NEG_INF) { alpha = 1.f; p0 = 0.f; p1 = 0.f; }  // fully masked so far
            else {
                alpha = __expf(m_i[r] - mn);      // exp(-inf)=0 handles first hit
                p0 = __expf(v0 - mn);
                p1 = __expf(v1 - mn);
            }
            float rs = p0 + p1;
            #pragma unroll
            for (int m = 1; m < 16; m <<= 1) rs += __shfl_xor(rs, m);
            l_i[r] = l_i[r] * alpha + rs;
            m_i[r] = mn;
            accv[0][r] *= alpha; accv[1][r] *= alpha;
            accv[2][r] *= alpha; accv[3][r] *= alpha;
            Ps[wave][r + 8 * ls][lm]      = (bf16_t)p0;
            Ps[wave][r + 8 * ls][16 + lm] = (bf16_t)p1;
        }

        // P(16x32) @ V(32x64): relayout P through this wave's private LDS tile.
        // A-fragment = two contiguous 8-half runs per lane -> 16B vector loads.
        v16bf pa = cat8(*(const v8bf*)&Ps[wave][lm][16 * ls],
                        *(const v8bf*)&Ps[wave][lm][16 * ls + 8]);
        #pragma unroll
        for (int j = 0; j < 4; ++j) {
            v16bf bvf;
            #pragma unroll
            for (int tt = 0; tt < 16; ++tt)
                bvf[tt] = Vs[cur][tt + 16 * ls][h * DH + j * 16 + lm];
            accv[j] = __builtin_amdgcn_wmma_f32_16x16x32_bf16(
                false, pa, false, bvf, (short)0, accv[j], false, false);
        }

#if USE_ASYNC_LDS
        __builtin_amdgcn_s_wait_asynccnt(0);      // prefetched chunk landed
#endif
        __syncthreads();
    }

    #pragma unroll
    for (int r = 0; r < 8; ++r) {
        int gn = rowBase + r + 8 * ls;
        float inv = 1.f / l_i[r];                 // self-loops guarantee l_i > 0
        #pragma unroll
        for (int j = 0; j < 4; ++j)
            hbuf[gn * DIM + h * DH + j * 16 + lm] = accv[j][r] * inv;
    }
}

// ---------------------------------------------------------------------------
// Kernel 3: out @ Wo + bo + residual(q_proj), then fused LayerNorm.
// ---------------------------------------------------------------------------
__global__ __launch_bounds__(256) void out_ln_kernel(
    const float* __restrict__ hbuf, const float* __restrict__ Wo,
    const float* __restrict__ bo,   const float* __restrict__ qproj,
    const float* __restrict__ gamma,const float* __restrict__ beta,
    float* __restrict__ out)
{
    __shared__ float hs[16][260];
    __shared__ float ys[16][257];
    const int t = threadIdx.x;
    const int rowBase = blockIdx.x * 16;
    for (int i = t; i < 16 * 256; i += 256) {
        int r = i >> 8, c = i & 255;
        hs[r][c] = hbuf[(rowBase + r) * DIM + c];
    }
    __syncthreads();

    const int wave = t >> 5, lane = t & 31, lm = lane & 15, ls = lane >> 4;
    for (int q = 0; q < 2; ++q) {
        int colBase = (wave * 2 + q) * 16;
        v8f acc = {0.f,0.f,0.f,0.f,0.f,0.f,0.f,0.f};
        #pragma unroll 8
        for (int k0 = 0; k0 < DIM; k0 += 4) {
            int k = k0 + 2 * ls;
            v2f a, b;
            a.x = hs[lm][k];
            a.y = hs[lm][k + 1];
            b.x = Wo[k * DIM + colBase + lm];
            b.y = Wo[(k + 1) * DIM + colBase + lm];
            acc = __builtin_amdgcn_wmma_f32_16x16x4_f32(
                false, a, false, b, (short)0, acc, false, false);
        }
        int col = colBase + lm;
        float bb = bo[col];
        #pragma unroll
        for (int r = 0; r < 8; ++r) {
            int lr = r + 8 * ls;
            ys[lr][col] = acc[r] + bb + qproj[(rowBase + lr) * DIM + col];
        }
    }
    __syncthreads();

    // LayerNorm: 16 threads per row (threads r*16+j live in one 16-lane half)
    int r = t >> 4, j = t & 15;
    float sum = 0.f;
    for (int c = j; c < DIM; c += 16) sum += ys[r][c];
    #pragma unroll
    for (int m = 1; m < 16; m <<= 1) sum += __shfl_xor(sum, m);
    float mean = sum * (1.f / 256.f);
    float vs = 0.f;
    for (int c = j; c < DIM; c += 16) { float d = ys[r][c] - mean; vs += d * d; }
    #pragma unroll
    for (int m = 1; m < 16; m <<= 1) vs += __shfl_xor(vs, m);
    float rstd = rsqrtf(vs * (1.f / 256.f) + 1e-5f);
    for (int c = j; c < DIM; c += 16)
        out[(rowBase + r) * DIM + c] = (ys[r][c] - mean) * rstd * gamma[c] + beta[c];
}

// ---------------------------------------------------------------------------
extern "C" void kernel_launch(void* const* d_in, const int* in_sizes, int n_in,
                              void* d_out, int out_size, void* d_ws, size_t ws_size,
                              hipStream_t stream) {
    const float* x     = (const float*)d_in[0];
    const int*   adj   = (const int*)  d_in[1];
    const float* Wq    = (const float*)d_in[2];
    const float* bq    = (const float*)d_in[3];
    const float* Wk    = (const float*)d_in[4];
    const float* bk    = (const float*)d_in[5];
    const float* Wv    = (const float*)d_in[6];
    const float* bv    = (const float*)d_in[7];
    const float* Wo    = (const float*)d_in[8];
    const float* bo    = (const float*)d_in[9];
    const float* gamma = (const float*)d_in[10];
    const float* beta  = (const float*)d_in[11];
    float* out = (float*)d_out;

    const size_t ND = (size_t)N_NODES * DIM;
    float*  qproj = (float*)d_ws;                 // 4 MB
    float*  hbuf  = qproj + ND;                   // 4 MB
    bf16_t* Qb    = (bf16_t*)(hbuf + ND);         // 2 MB
    bf16_t* Kb    = Qb + ND;                      // 2 MB
    bf16_t* Vb    = Kb + ND;                      // 2 MB (14 MB total)

    qkv_kernel   <<<N_NODES / 16, 256, 0, stream>>>(x, Wq, bq, Wk, bk, Wv, bv,
                                                    qproj, Qb, Kb, Vb);
    attn_kernel  <<<N_NODES / 32, 256, 0, stream>>>(Qb, Kb, Vb, adj, hbuf);
    out_ln_kernel<<<N_NODES / 16, 256, 0, stream>>>(hbuf, Wo, bo, qproj,
                                                    gamma, beta, out);
}
